// CellLSTM_84774064489112
// MI455X (gfx1250) — matrix operand order repaired and checked by reading, not compiled, not run on hardware
//
#include <hip/hip_runtime.h>

typedef __bf16 bf16;
typedef __attribute__((ext_vector_type(16))) __bf16 v16bf;
typedef __attribute__((ext_vector_type(8)))  float  v8f;

#define Bc   64
#define Tc   512
#define Ic   256
#define Hc   768
#define G4H  3072   // 4*H

// ---------------------------------------------------------------------------
// Fragment loaders (layouts per CDNA5 ISA 7.12.2, wave32)
// A 16x32 bf16 (MxK): lane l -> M = l&15; halves 0..7  = K0+ (l>>4)*8 + 0..7
//                                      halves 8..15 = K0+16+(l>>4)*8 + 0..7
// B 32x16 bf16 (KxN): lane l -> N = l&15; same K mapping as A
// C/D 16x16 f32:      lane l, vgpr r -> M = r + (l>>4)*8, N = l&15
// ---------------------------------------------------------------------------
union ABFrag { v16bf v; uint4 q[2]; };

__device__ __forceinline__ v16bf load_frag_a(const bf16* rowk0, int hi) {
  ABFrag u;
  const char* p = (const char*)rowk0 + hi * 16;   // hi*8 halves
  u.q[0] = *(const uint4*)(p);
  u.q[1] = *(const uint4*)(p + 32);               // +16 halves
  return u.v;
}

__device__ __forceinline__ v16bf load_frag_b(const bf16* rowk0, int hi) {
  ABFrag u;
  const char* p = (const char*)rowk0 + hi * 16;   // hi*8 halves
  u.q[0] = *(const uint4*)(p);
  u.q[1] = *(const uint4*)(p + 32);
  return u.v;
}

__device__ __forceinline__ float sigmoidf_(float x) {
  return 1.0f / (1.0f + __expf(-x));
}

// ---------------------------------------------------------------------------
// Device-wide sense barrier (24 co-resident WGs; MI455X has far more WGPs)
// ---------------------------------------------------------------------------
__device__ __forceinline__ void grid_sync(int* bar, int nwg) {
  __syncthreads();
  if (threadIdx.x == 0) {
    int gen = __hip_atomic_load(&bar[1], __ATOMIC_RELAXED, __HIP_MEMORY_SCOPE_AGENT);
    int old = __hip_atomic_fetch_add(&bar[0], 1, __ATOMIC_ACQ_REL, __HIP_MEMORY_SCOPE_AGENT);
    if (old == nwg - 1) {
      __hip_atomic_store(&bar[0], 0, __ATOMIC_RELAXED, __HIP_MEMORY_SCOPE_AGENT);
      __hip_atomic_fetch_add(&bar[1], 1, __ATOMIC_ACQ_REL, __HIP_MEMORY_SCOPE_AGENT);
    } else {
      while (__hip_atomic_load(&bar[1], __ATOMIC_ACQUIRE, __HIP_MEMORY_SCOPE_AGENT) == gen)
        __builtin_amdgcn_s_sleep(1);
    }
  }
  __syncthreads();
}

// ---------------------------------------------------------------------------
// fp32 -> bf16 convert
// ---------------------------------------------------------------------------
__global__ void f32_to_bf16(const float* __restrict__ in, bf16* __restrict__ out, int n) {
  int i = blockIdx.x * blockDim.x + threadIdx.x;
  int stride = gridDim.x * blockDim.x;
  for (; i < n; i += stride) out[i] = (bf16)in[i];
}

__global__ void init_bar(int* bar) { bar[0] = 0; bar[1] = 0; }

// ---------------------------------------------------------------------------
// gx = A(bf16, MxK row-major) @ W(bf16, 3072xK row-major)^T + (b_ih + b_hh)
// Grid: (N/256, M/64), 256 threads. Wave tile 32x64 (2 Mtiles x 4 Ntiles).
// ---------------------------------------------------------------------------
__global__ __launch_bounds__(256) void gemm_gx(
    const bf16* __restrict__ A, const bf16* __restrict__ W,
    const float* __restrict__ bih, const float* __restrict__ bhh,
    float* __restrict__ gxout, int K)
{
  const int tid = threadIdx.x, wave = tid >> 5, lane = tid & 31;
  const int hi = lane >> 4, ln = lane & 15;
  const int n0 = blockIdx.x * 256 + (wave >> 1) * 64;
  const int m0 = blockIdx.y * 64  + (wave & 1)  * 32;

  v8f acc[2][4] = {};
  const bf16* arow0 = A + (size_t)(m0 + ln)      * K;
  const bf16* arow1 = A + (size_t)(m0 + 16 + ln) * K;
  const bf16* brow[4];
#pragma unroll
  for (int nt = 0; nt < 4; ++nt) brow[nt] = W + (size_t)(n0 + nt * 16 + ln) * K;

  for (int k0 = 0; k0 < K; k0 += 32) {
    v16bf a0 = load_frag_a(arow0 + k0, hi);
    v16bf a1 = load_frag_a(arow1 + k0, hi);
#pragma unroll
    for (int nt = 0; nt < 4; ++nt) {
      v16bf b = load_frag_b(brow[nt] + k0, hi);
      acc[0][nt] = __builtin_amdgcn_wmma_f32_16x16x32_bf16(false, a0, false, b,
                     (short)0, acc[0][nt], false, false);
      acc[1][nt] = __builtin_amdgcn_wmma_f32_16x16x32_bf16(false, a1, false, b,
                     (short)0, acc[1][nt], false, false);
    }
  }

#pragma unroll
  for (int nt = 0; nt < 4; ++nt) {
    int col = n0 + nt * 16 + ln;
    float bsum = bih[col] + bhh[col];
#pragma unroll
    for (int mt = 0; mt < 2; ++mt) {
      int mrow = m0 + mt * 16 + hi * 8;
#pragma unroll
      for (int r = 0; r < 8; ++r)
        gxout[(size_t)(mrow + r) * G4H + col] = acc[mt][nt][r] + bsum;
    }
  }
}

// ---------------------------------------------------------------------------
// Persistent LSTM scan. 24 WGs x 256 thr. WG wg owns hidden units [wg*32, +32):
// its 128 gate rows of w_hh (i,f,g,o blocks of 32) live in LDS (pitch-padded).
// Wave (of 8): mtile = w&3 (batch 16-slice), ht = w>>2 (hidden 16-slice);
// 4 accumulators = same (m,j) tile of gates i/f/g/o -> gate math is lane-local.
// c state stays in registers for all 512 steps.
// ---------------------------------------------------------------------------
#define LPITCH 776   // 768 + 8 halves pad (bank spread), 16B-aligned rows

__global__ __launch_bounds__(256) void lstm_scan(
    const float* __restrict__ gx,   // (B*T) x 4H fp32, row = b*T + t
    const bf16*  __restrict__ whh,  // 4H x H bf16 row-major
    bf16* __restrict__ hist,        // (B*T) x H bf16: h history (row = b*T + t)
    float* ys,                      // optional fp32 ys out (same row layout)
    float* __restrict__ hout, float* __restrict__ cout,
    int* bar)
{
  extern __shared__ char smem_raw[];
  bf16* lw = (bf16*)smem_raw;

  const int tid = threadIdx.x;
  const int j0  = blockIdx.x * 32;

  // Stage this WG's 128 w_hh rows into LDS (192KB payload, one-time cost)
  for (int idx = tid; idx < 128 * 96; idx += 256) {
    int r = idx / 96, ch = idx % 96;          // 96 x 16B per row
    int gate = r >> 5, jj = r & 31;
    const uint4* src = (const uint4*)(whh + (size_t)(gate * Hc + j0 + jj) * Hc) + ch;
    uint4* dst = (uint4*)(lw + (size_t)r * LPITCH) + ch;
    *dst = *src;
  }
  __syncthreads();

  const int wave = tid >> 5, lane = tid & 31;
  const int hi = lane >> 4, ln = lane & 15;
  const int mtile = wave & 3, ht = wave >> 2;
  const int m_a    = mtile * 16 + ln;        // A-matrix row (batch) this lane loads
  const int m_base = mtile * 16 + hi * 8;    // C/D rows (batch) this lane holds
  const int jl = ht * 16 + ln;               // local hidden 0..31
  const int j  = j0 + jl;                    // global hidden

  const bf16* brow[4];
#pragma unroll
  for (int g = 0; g < 4; ++g) brow[g] = lw + (size_t)(g * 32 + jl) * LPITCH;

  v8f c = {};
  const bool wy = (ys != nullptr);

  for (int t = 0; t < Tc; ++t) {
    v8f acc[4] = {};
    if (t > 0) {
      const bf16* ar = hist + ((size_t)m_a * Tc + (t - 1)) * Hc;
#pragma unroll 4
      for (int k0 = 0; k0 < Hc; k0 += 32) {
        v16bf a = load_frag_a(ar + k0, hi);
#pragma unroll
        for (int g = 0; g < 4; ++g) {
          v16bf b = load_frag_b(brow[g] + k0, hi);
          acc[g] = __builtin_amdgcn_wmma_f32_16x16x32_bf16(false, a, false, b,
                     (short)0, acc[g], false, false);
        }
      }
    }

    float hv[8];
#pragma unroll
    for (int r = 0; r < 8; ++r) {
      size_t gxr = ((size_t)(m_base + r) * Tc + t) * (size_t)G4H + j;
      float gi = acc[0][r] + gx[gxr];
      float gf = acc[1][r] + gx[gxr + 768];
      float gg = acc[2][r] + gx[gxr + 1536];
      float go = acc[3][r] + gx[gxr + 2304];
      gi = sigmoidf_(gi);
      gf = sigmoidf_(gf);
      gg = tanhf(gg);
      go = sigmoidf_(go);
      float cv = gf * c[r] + gi * gg;
      c[r] = cv;
      hv[r] = go * tanhf(cv);
    }

#pragma unroll
    for (int r = 0; r < 8; ++r) {
      size_t row = (size_t)(m_base + r) * Tc + t;
      hist[row * Hc + j] = (bf16)hv[r];
      if (wy) ys[row * Hc + j] = hv[r];
    }
    if (t == Tc - 1) {
#pragma unroll
      for (int r = 0; r < 8; ++r) {
        hout[(size_t)(m_base + r) * Hc + j] = hv[r];
        cout[(size_t)(m_base + r) * Hc + j] = c[r];
      }
    }
    grid_sync(bar, gridDim.x);   // publish h(t) before anyone reads it at t+1
  }
}

// ---------------------------------------------------------------------------
// Host side
// ---------------------------------------------------------------------------
extern "C" void kernel_launch(void* const* d_in, const int* in_sizes, int n_in,
                              void* d_out, int out_size, void* d_ws, size_t ws_size,
                              hipStream_t stream) {
  const float* x    = (const float*)d_in[0];
  const float* wih0 = (const float*)d_in[1];
  const float* whh0 = (const float*)d_in[2];
  const float* bih0 = (const float*)d_in[3];
  const float* bhh0 = (const float*)d_in[4];
  const float* wih1 = (const float*)d_in[5];
  const float* whh1 = (const float*)d_in[6];
  const float* bih1 = (const float*)d_in[7];
  const float* bhh1 = (const float*)d_in[8];

  char* ws = (char*)d_ws;
  float* gx    = (float*)(ws);                        // 32768*3072*4 = 402,653,184
  bf16*  xbf   = (bf16*)(ws + 402653184ULL);          // 16,777,216
  bf16*  wihb0 = (bf16*)(ws + 419430400ULL);          // 1,572,864
  bf16*  whhb0 = (bf16*)(ws + 421003264ULL);          // 4,718,592
  bf16*  wihb1 = (bf16*)(ws + 425721856ULL);          // 4,718,592
  bf16*  whhb1 = (bf16*)(ws + 430440448ULL);          // 4,718,592
  bf16*  ysbf  = (bf16*)(ws + 435159040ULL);          // 50,331,648 (shared h history)
  int*   bar   = (int*) (ws + 485490688ULL);

  float* out = (float*)d_out;
  float* ys  = out;                    // 64*512*768
  float* h0  = out + 25165824;
  float* h1  = out + 25214976;
  float* c0  = out + 25264128;
  float* c1  = out + 25313280;

  hipLaunchKernelGGL(init_bar, dim3(1), dim3(1), 0, stream, bar);
  hipLaunchKernelGGL(f32_to_bf16, dim3(512), dim3(256), 0, stream, x,    xbf,   Bc*Tc*Ic);
  hipLaunchKernelGGL(f32_to_bf16, dim3(64),  dim3(256), 0, stream, wih0, wihb0, G4H*Ic);
  hipLaunchKernelGGL(f32_to_bf16, dim3(128), dim3(256), 0, stream, whh0, whhb0, G4H*Hc);
  hipLaunchKernelGGL(f32_to_bf16, dim3(128), dim3(256), 0, stream, wih1, wihb1, G4H*Hc);
  hipLaunchKernelGGL(f32_to_bf16, dim3(128), dim3(256), 0, stream, whh1, whhb1, G4H*Hc);

  const size_t ldsBytes = 128ULL * LPITCH * sizeof(bf16);   // 198,656 B (<320KB/WGP)

  // Layer 0
  hipLaunchKernelGGL(gemm_gx, dim3(12, 512), dim3(256), 0, stream,
                     xbf, wihb0, bih0, bhh0, gx, Ic);
  hipLaunchKernelGGL(lstm_scan, dim3(24), dim3(256), ldsBytes, stream,
                     gx, whhb0, ysbf, (float*)nullptr, h0, c0, bar);
  // Layer 1 (ysbf = layer-0 output feeds the GEMM, then becomes layer-1 history)
  hipLaunchKernelGGL(gemm_gx, dim3(12, 512), dim3(256), 0, stream,
                     ysbf, wihb1, bih1, bhh1, gx, Hc);
  hipLaunchKernelGGL(lstm_scan, dim3(24), dim3(256), ldsBytes, stream,
                     gx, whhb1, ysbf, ys, h1, c1, bar);
}
